// MultiHeadSelfAttention_65120294142355
// MI455X (gfx1250) — matrix operand order repaired and checked by reading, not compiled, not run on hardware
//
#include <hip/hip_runtime.h>

typedef __bf16 bf16;
typedef __attribute__((ext_vector_type(16))) __bf16 v16bf;
typedef __attribute__((ext_vector_type(8)))  __bf16 v8bf;
typedef __attribute__((ext_vector_type(8)))  float  v8f;

#define BDIM  4
#define SDIM  2048
#define DDIM  512
#define HDIM  8
#define DEPTH 64
#define NROW  (BDIM * SDIM)   // 8192

// ---------------------------------------------------------------------------
// WMMA helpers (CDNA5 layouts, wave32)
// ---------------------------------------------------------------------------
__device__ __forceinline__ v8f wmma_bf16(v16bf a, v16bf b, v8f c) {
    return __builtin_amdgcn_wmma_f32_16x16x32_bf16(
        /*neg_a=*/false, a, /*neg_b=*/false, b,
        /*c_mod=*/(short)0, c, /*reuse_a=*/false, /*reuse_b=*/false);
}

// A-matrix fragment, 16x32 bf16, from row-major [rows x lda].
// lanes 0-15: M=lane, K = kBase+{0..7, 16..23}; lanes 16-31: K = kBase+{8..15, 24..31}
__device__ __forceinline__ v16bf load_a_frag(const bf16* A, int lda,
                                             int rowBase, int kBase, int lane) {
    const int m   = lane & 15;
    const int klo = (lane < 16) ? 0 : 8;
    const bf16* p = A + (size_t)(rowBase + m) * lda + kBase + klo;
    v8bf lo = *(const v8bf*)(p);
    v8bf hi = *(const v8bf*)(p + 16);
    v16bf r;
#pragma unroll
    for (int i = 0; i < 8; ++i) { r[i] = lo[i]; r[i + 8] = hi[i]; }
    return r;
}

// B-matrix fragment, 32x16 bf16, where B(k,n) is stored transposed: Bt[n][k], ld = ldb.
// lane = column n (mod 16); lanes 0-15 hold K = kBase+{0..7, 16..23}, lanes 16-31 K = kBase+{8..15, 24..31}
__device__ __forceinline__ v16bf load_bt_frag(const bf16* Bt, int ldb,
                                              int nBase, int kBase, int lane) {
    const int n    = lane & 15;
    const int koff = (lane < 16) ? 0 : 8;
    const bf16* p  = Bt + (size_t)(nBase + n) * ldb + kBase + koff;
    v8bf lo = *(const v8bf*)(p);
    v8bf hi = *(const v8bf*)(p + 16);
    v16bf r;
#pragma unroll
    for (int i = 0; i < 8; ++i) { r[i] = lo[i]; r[i + 8] = hi[i]; }
    return r;
}

// Async Global->LDS copy of 16 bytes per lane (CDNA5, ASYNCcnt-tracked).
// LDS operand = low 32 bits of the generic pointer (HW truncates generic->LDS).
__device__ __forceinline__ void async_copy16(const bf16* gsrc, bf16* lds_dst) {
    unsigned laddr = (unsigned)(unsigned long long)lds_dst;
    asm volatile("global_load_async_to_lds_b128 %0, %1, off"
                 :: "v"(laddr), "v"(gsrc)
                 : "memory");
}
__device__ __forceinline__ void wait_async0() {
    asm volatile("s_wait_asynccnt 0" ::: "memory");
}

// ---------------------------------------------------------------------------
// Conversion kernels
// ---------------------------------------------------------------------------
__global__ void k_cvt(const float* __restrict__ src, bf16* __restrict__ dst, int n) {
    int i = blockIdx.x * blockDim.x + threadIdx.x;
    if (i < n) dst[i] = (bf16)src[i];
}

// w: row-major [k][n] (512x512)  ->  wt: [n][k]  (bf16)
__global__ void k_cvt_t(const float* __restrict__ w, bf16* __restrict__ wt) {
    int i  = blockIdx.x * blockDim.x + threadIdx.x;   // over 512*512
    int kk = i >> 9;
    int n  = i & 511;
    wt[(size_t)n * DDIM + kk] = (bf16)w[i];
}

// ---------------------------------------------------------------------------
// QKV projection: [8192,512] @ [512,512] x3.  One wave = 16 rows x 64 cols.
// Block = 8 waves = 32 rows x 256 cols. grid = (2, 256, 3)
// Q,K -> [B,H,S,64]; V -> [B,H,64,S] (transposed for P@V B-operand)
// ---------------------------------------------------------------------------
__global__ void k_gemm_qkv(const bf16* __restrict__ xb,
                           const bf16* __restrict__ wqt, const bf16* __restrict__ wkt,
                           const bf16* __restrict__ wvt,
                           bf16* __restrict__ qb, bf16* __restrict__ kb,
                           bf16* __restrict__ vt) {
    const int lane    = threadIdx.x & 31;
    const int w       = threadIdx.x >> 5;
    const int rowBase = blockIdx.y * 32 + (w & 1) * 16;
    const int colBase = blockIdx.x * 256 + (w >> 1) * 64;
    const int mode    = blockIdx.z;
    const bf16* wt = (mode == 0) ? wqt : (mode == 1) ? wkt : wvt;

    v8f acc[4];
#pragma unroll
    for (int c = 0; c < 4; ++c)
#pragma unroll
        for (int i = 0; i < 8; ++i) acc[c][i] = 0.0f;

    for (int k = 0; k < DDIM; k += 32) {
        v16bf a = load_a_frag(xb, DDIM, rowBase, k, lane);
#pragma unroll
        for (int c = 0; c < 4; ++c) {
            v16bf b = load_bt_frag(wt, DDIM, colBase + c * 16, k, lane);
            acc[c] = wmma_bf16(a, b, acc[c]);
        }
    }

    const int hi8  = (lane < 16) ? 0 : 8;
    const int col0 = lane & 15;
#pragma unroll
    for (int c = 0; c < 4; ++c) {
#pragma unroll
        for (int r = 0; r < 8; ++r) {
            int row = rowBase + r + hi8;           // global row = b*S + s
            int b_  = row >> 11;
            int s   = row & (SDIM - 1);
            int col = colBase + c * 16 + col0;
            int h   = col >> 6;
            int dd  = col & 63;
            bf16 v = (bf16)acc[c][r];
            if (mode == 2)
                vt[(((size_t)b_ * HDIM + h) * DEPTH + dd) * SDIM + s] = v;
            else if (mode == 0)
                qb[(((size_t)b_ * HDIM + h) * SDIM + s) * DEPTH + dd] = v;
            else
                kb[(((size_t)b_ * HDIM + h) * SDIM + s) * DEPTH + dd] = v;
        }
    }
}

// ---------------------------------------------------------------------------
// Flash attention. Block = 8 waves, all on the same (b,h), 8 consecutive
// 16-query tiles (128 queries). K/V tiles (32 keys) are staged once per block
// into double-buffered LDS via async Global->LDS DMA and shared by all waves.
// Per key tile: 4 WMMA (QK^T) + online softmax + 4 WMMA (P@V).
// ---------------------------------------------------------------------------
__global__ void k_attn(const bf16* __restrict__ qb, const bf16* __restrict__ kb,
                       const bf16* __restrict__ vt, bf16* __restrict__ ctxb) {
    __shared__ __bf16 ldsK[2][32 * DEPTH];  // [key_local][depth]   2 x 4 KiB
    __shared__ __bf16 ldsV[2][DEPTH * 32];  // [depth][key_local]   2 x 4 KiB
    __shared__ __bf16 ldsP[8][16 * 32];     // per-wave P bounce    8 KiB

    const int tid  = threadIdx.x;
    const int lane = tid & 31;
    const int w    = tid >> 5;

    const int bh    = blockIdx.x >> 4;        // 0..31  (b*8 + h)
    const int qblk  = blockIdx.x & 15;        // 16 blocks of 128 queries
    const int b     = bh >> 3;
    const int h     = bh & 7;
    const int qBase = qblk * 128 + w * 16;

    const bf16* qbh = qb + (size_t)bh * SDIM * DEPTH;
    const bf16* kbh = kb + (size_t)bh * SDIM * DEPTH;
    const bf16* vbh = vt + (size_t)bh * DEPTH * SDIM;

    // per-thread source/dest for the staged tiles (256 threads x 16 B = 4 KiB each)
    const int krow   = tid >> 3;              // 0..31 key rows
    const int kchunk = (tid & 7) * 8;         // 8 bf16 per chunk
    const int vrow   = tid >> 2;              // 0..63 depth rows
    const int vchunk = (tid & 3) * 8;

    v16bf q0 = load_a_frag(qbh, DEPTH, qBase, 0, lane);
    v16bf q1 = load_a_frag(qbh, DEPTH, qBase, 32, lane);

    float m[8], l[8];
    v8f ctx[4];
#pragma unroll
    for (int r = 0; r < 8; ++r) { m[r] = -1e30f; l[r] = 0.0f; }
#pragma unroll
    for (int c = 0; c < 4; ++c)
#pragma unroll
        for (int i = 0; i < 8; ++i) ctx[c][i] = 0.0f;

    __bf16* myP = &ldsP[w][0];
    const int hi8  = (lane < 16) ? 0 : 8;
    const int col0 = lane & 15;

    // prologue: stage tile 0
    async_copy16(kbh + (size_t)krow * DEPTH + kchunk, &ldsK[0][krow * DEPTH + kchunk]);
    async_copy16(vbh + (size_t)vrow * SDIM + vchunk,  &ldsV[0][vrow * 32 + vchunk]);
    wait_async0();
    __syncthreads();

    int cur = 0;
    for (int j = 0; j < SDIM; j += 32) {
        const int nxt = cur ^ 1;
        if (j + 32 < SDIM) {   // prefetch next K/V tile into the other buffer
            async_copy16(kbh + (size_t)(j + 32 + krow) * DEPTH + kchunk,
                         &ldsK[nxt][krow * DEPTH + kchunk]);
            async_copy16(vbh + (size_t)vrow * SDIM + (j + 32) + vchunk,
                         &ldsV[nxt][vrow * 32 + vchunk]);
        }
        const bf16* Kc = &ldsK[cur][0];
        const bf16* Vc = &ldsV[cur][0];

        // ---- logits: 16 queries x 32 keys, depth 64 (from LDS) ----
        v8f s0, s1;
#pragma unroll
        for (int i = 0; i < 8; ++i) { s0[i] = 0.0f; s1[i] = 0.0f; }
        {
            v16bf kf;
            kf = load_bt_frag(Kc, DEPTH, 0,  0,  lane); s0 = wmma_bf16(q0, kf, s0);
            kf = load_bt_frag(Kc, DEPTH, 0,  32, lane); s0 = wmma_bf16(q1, kf, s0);
            kf = load_bt_frag(Kc, DEPTH, 16, 0,  lane); s1 = wmma_bf16(q0, kf, s1);
            kf = load_bt_frag(Kc, DEPTH, 16, 32, lane); s1 = wmma_bf16(q1, kf, s1);
        }

        // ---- scale + row max (reduce over 16 lanes of each half) ----
        float rm[8];
#pragma unroll
        for (int r = 0; r < 8; ++r) {
            s0[r] *= 0.125f;                 // 1/sqrt(64)
            s1[r] *= 0.125f;
            rm[r] = fmaxf(s0[r], s1[r]);
        }
#pragma unroll
        for (int off = 1; off < 16; off <<= 1)
#pragma unroll
            for (int r = 0; r < 8; ++r)
                rm[r] = fmaxf(rm[r], __shfl_xor(rm[r], off, 32));

        // ---- online softmax update ----
        float alpha[8], rs[8];
#pragma unroll
        for (int r = 0; r < 8; ++r) {
            float nm = fmaxf(m[r], rm[r]);
            alpha[r] = __expf(m[r] - nm);
            m[r]     = nm;
        }
#pragma unroll
        for (int c = 0; c < 4; ++c)
#pragma unroll
            for (int r = 0; r < 8; ++r) ctx[c][r] *= alpha[r];
#pragma unroll
        for (int r = 0; r < 8; ++r) {
            s0[r] = __expf(s0[r] - m[r]);
            s1[r] = __expf(s1[r] - m[r]);
            rs[r] = s0[r] + s1[r];
        }
#pragma unroll
        for (int off = 1; off < 16; off <<= 1)
#pragma unroll
            for (int r = 0; r < 8; ++r) rs[r] += __shfl_xor(rs[r], off, 32);
#pragma unroll
        for (int r = 0; r < 8; ++r) l[r] = l[r] * alpha[r] + rs[r];

        // ---- C-layout P -> A-fragment via per-wave LDS bounce ----
#pragma unroll
        for (int r = 0; r < 8; ++r) {
            myP[(r + hi8) * 32 + col0]      = (bf16)s0[r];
            myP[(r + hi8) * 32 + 16 + col0] = (bf16)s1[r];
        }
        asm volatile("s_wait_dscnt 0" ::: "memory");
        v16bf pf;
        {
            const int mm  = lane & 15;
            const int klo = (lane < 16) ? 0 : 8;
            v8bf lo = *(const v8bf*)&myP[mm * 32 + klo];
            v8bf hi = *(const v8bf*)&myP[mm * 32 + klo + 16];
#pragma unroll
            for (int i = 0; i < 8; ++i) { pf[i] = lo[i]; pf[i + 8] = hi[i]; }
        }

        // ---- ctx += P @ V  (V tile in LDS as [depth][32]) ----
#pragma unroll
        for (int c = 0; c < 4; ++c) {
            v16bf vf = load_bt_frag(Vc, 32, c * 16, 0, lane);
            ctx[c] = wmma_bf16(pf, vf, ctx[c]);
        }

        // retire prefetch + release current buffer for overwrite next iter
        wait_async0();
        __syncthreads();
        cur = nxt;
    }

    // ---- normalize and store ctx as bf16 into [B*S, 512] ----
    float inv[8];
#pragma unroll
    for (int r = 0; r < 8; ++r) inv[r] = 1.0f / l[r];
#pragma unroll
    for (int c = 0; c < 4; ++c)
#pragma unroll
        for (int r = 0; r < 8; ++r) {
            int row = qBase + r + hi8;
            int col = h * 64 + c * 16 + col0;
            ctxb[((size_t)b * SDIM + row) * DDIM + col] = (bf16)(ctx[c][r] * inv[r]);
        }
}

// ---------------------------------------------------------------------------
// Output projection fused with bias + residual: resid = x + ctx@wo + bo
// ---------------------------------------------------------------------------
__global__ void k_gemm_out(const bf16* __restrict__ ctxb, const bf16* __restrict__ wot,
                           const float* __restrict__ x, const float* __restrict__ bo,
                           float* __restrict__ resid) {
    const int lane    = threadIdx.x & 31;
    const int w       = threadIdx.x >> 5;
    const int rowBase = blockIdx.y * 32 + (w & 1) * 16;
    const int colBase = blockIdx.x * 256 + (w >> 1) * 64;

    v8f acc[4];
#pragma unroll
    for (int c = 0; c < 4; ++c)
#pragma unroll
        for (int i = 0; i < 8; ++i) acc[c][i] = 0.0f;

    for (int k = 0; k < DDIM; k += 32) {
        v16bf a = load_a_frag(ctxb, DDIM, rowBase, k, lane);
#pragma unroll
        for (int c = 0; c < 4; ++c) {
            v16bf b = load_bt_frag(wot, DDIM, colBase + c * 16, k, lane);
            acc[c] = wmma_bf16(a, b, acc[c]);
        }
    }

    const int hi8  = (lane < 16) ? 0 : 8;
    const int col0 = lane & 15;
#pragma unroll
    for (int c = 0; c < 4; ++c) {
#pragma unroll
        for (int r = 0; r < 8; ++r) {
            int row = rowBase + r + hi8;
            int col = colBase + c * 16 + col0;
            size_t idx = (size_t)row * DDIM + col;
            resid[idx] = x[idx] + acc[c][r] + bo[col];
        }
    }
}

// ---------------------------------------------------------------------------
// LayerNorm over D=512 per row; one wave per row, 8 rows per block
// ---------------------------------------------------------------------------
__global__ void k_layernorm(const float* __restrict__ resid, const float* __restrict__ gamma,
                            const float* __restrict__ beta, float* __restrict__ out) {
    const int lane = threadIdx.x & 31;
    const int w    = threadIdx.x >> 5;
    const int row  = blockIdx.x * 8 + w;
    const float* rp = resid + (size_t)row * DDIM;

    float vals[16];
    float s = 0.0f, ss = 0.0f;
#pragma unroll
    for (int i = 0; i < 16; ++i) {
        float v = rp[lane + i * 32];
        vals[i] = v;
        s += v;
        ss += v * v;
    }
#pragma unroll
    for (int off = 1; off < 32; off <<= 1) {
        s  += __shfl_xor(s, off, 32);
        ss += __shfl_xor(ss, off, 32);
    }
    float mu   = s * (1.0f / 512.0f);
    float var  = ss * (1.0f / 512.0f) - mu * mu;
    float rstd = rsqrtf(var + 1e-6f);
#pragma unroll
    for (int i = 0; i < 16; ++i) {
        int c = lane + i * 32;
        out[(size_t)row * DDIM + c] = (vals[i] - mu) * rstd * gamma[c] + beta[c];
    }
}

// ---------------------------------------------------------------------------
// Launch
// ---------------------------------------------------------------------------
extern "C" void kernel_launch(void* const* d_in, const int* in_sizes, int n_in,
                              void* d_out, int out_size, void* d_ws, size_t ws_size,
                              hipStream_t stream) {
    const float* x     = (const float*)d_in[0];
    const float* wq    = (const float*)d_in[1];
    const float* wk    = (const float*)d_in[2];
    const float* wv    = (const float*)d_in[3];
    const float* wo    = (const float*)d_in[4];
    const float* bo    = (const float*)d_in[5];
    const float* gamma = (const float*)d_in[6];
    const float* beta  = (const float*)d_in[7];
    float* out = (float*)d_out;

    char* ws = (char*)d_ws;
    constexpr size_t XB_SZ  = (size_t)NROW * DDIM * 2;   // 8 MiB (bf16 [8192,512])
    constexpr size_t WT_SZ  = (size_t)DDIM * DDIM * 2;   // 512 KiB each
    constexpr size_t XB_OFF  = 0;
    constexpr size_t WQT_OFF = XB_OFF + XB_SZ;
    constexpr size_t WKT_OFF = WQT_OFF + WT_SZ;
    constexpr size_t WVT_OFF = WKT_OFF + WT_SZ;
    constexpr size_t WOT_OFF = WVT_OFF + WT_SZ;
    constexpr size_t QB_OFF  = WOT_OFF + WT_SZ;
    constexpr size_t KB_OFF  = QB_OFF + XB_SZ;
    constexpr size_t VT_OFF  = KB_OFF + XB_SZ;
    constexpr size_t CTX_OFF = VT_OFF + XB_SZ;
    constexpr size_t RES_OFF = CTX_OFF + XB_SZ;          // float [8192,512] = 16 MiB

    bf16* xb   = (bf16*)(ws + XB_OFF);
    bf16* wqt  = (bf16*)(ws + WQT_OFF);
    bf16* wkt  = (bf16*)(ws + WKT_OFF);
    bf16* wvt  = (bf16*)(ws + WVT_OFF);
    bf16* wot  = (bf16*)(ws + WOT_OFF);
    bf16* qb   = (bf16*)(ws + QB_OFF);
    bf16* kb   = (bf16*)(ws + KB_OFF);
    bf16* vt   = (bf16*)(ws + VT_OFF);
    bf16* ctxb = (bf16*)(ws + CTX_OFF);
    float* resid = (float*)(ws + RES_OFF);

    // 1) precision conversion / weight transposes
    k_cvt<<<(NROW * DDIM) / 256, 256, 0, stream>>>(x, xb, NROW * DDIM);
    k_cvt_t<<<(DDIM * DDIM) / 256, 256, 0, stream>>>(wq, wqt);
    k_cvt_t<<<(DDIM * DDIM) / 256, 256, 0, stream>>>(wk, wkt);
    k_cvt_t<<<(DDIM * DDIM) / 256, 256, 0, stream>>>(wv, wvt);
    k_cvt_t<<<(DDIM * DDIM) / 256, 256, 0, stream>>>(wo, wot);

    // 2) QKV projections (WMMA)
    dim3 gq(DDIM / 256, NROW / 32, 3);
    k_gemm_qkv<<<gq, 256, 0, stream>>>(xb, wqt, wkt, wvt, qb, kb, vt);

    // 3) flash attention (WMMA + async LDS staging + online softmax)
    k_attn<<<BDIM * HDIM * (SDIM / 128), 256, 0, stream>>>(qb, kb, vt, ctxb);

    // 4) output projection + bias + residual (WMMA)
    dim3 go(DDIM / 256, NROW / 32, 1);
    k_gemm_out<<<go, 256, 0, stream>>>(ctxb, wot, x, bo, resid);

    // 5) layernorm
    k_layernorm<<<NROW / 8, 256, 0, stream>>>(resid, gamma, beta, out);
}
